// BeltActorTransformer_51668456571280
// MI455X (gfx1250) — compile-verified, matrix operand order checked by reading, not run on hardware
//
#include <hip/hip_runtime.h>

typedef __attribute__((ext_vector_type(16))) _Float16 v16h;
typedef __attribute__((ext_vector_type(8)))  _Float16 v8h;
typedef __attribute__((ext_vector_type(8)))  float    v8f;
typedef __attribute__((ext_vector_type(4)))  int      v4i;

// ---------------------------------------------------------------------------
// Feature detection: gfx1250 async global->LDS DMA (ASYNCcnt path)
// ---------------------------------------------------------------------------
#if defined(__has_builtin)
#if __has_builtin(__builtin_amdgcn_global_load_async_to_lds_b128)
#define HAVE_ASYNC_LDS 1
#endif
#if __has_builtin(__builtin_amdgcn_s_wait_asynccnt)
#define HAVE_WAIT_ASYNC 1
#endif
#endif

#ifdef HAVE_ASYNC_LDS
#define AS1 __attribute__((address_space(1)))
#define AS3 __attribute__((address_space(3)))
__device__ inline void async_copy16(const void* g, void* l) {
  __builtin_amdgcn_global_load_async_to_lds_b128((AS1 v4i*)(void*)g,
                                                 (AS3 v4i*)l, 0, 0);
}
__device__ inline void wait_async0() {
#ifdef HAVE_WAIT_ASYNC
  __builtin_amdgcn_s_wait_asynccnt(0);
#else
  asm volatile("s_wait_asynccnt 0x0" ::: "memory");
#endif
}
#endif

// ---------------------------------------------------------------------------
// Model constants
// ---------------------------------------------------------------------------
#define D_MODEL 512
#define NHEAD   8
#define DH      64
#define NLAYERS 4
#define OBS_DIM 64
#define SEQ     256
#define KP1     128
#define NBATCH  64
#define NTOK    (NBATCH * SEQ)     // 16384
#define FFN     2048

// ---------------------------------------------------------------------------
// WMMA helpers (CDNA5 16x16x32 f16 -> f32)
// ---------------------------------------------------------------------------
__device__ inline v8f zero8() {
  v8f z;
#pragma unroll
  for (int i = 0; i < 8; ++i) z[i] = 0.f;
  return z;
}

__device__ inline v8f wmma_f16(v16h a, v16h b, v8f c) {
  return __builtin_amdgcn_wmma_f32_16x16x32_f16(false, a, false, b, (short)0, c,
                                                false, false);
}

// Load a 16x32 fragment (A layout, or B layout of W stored as [N][K]) from
// row-major f16 LDS/memory with row stride `stride` (in halves, must be even).
// Lane L (L&15 = row), elements per ISA 7.12.2:
//   j<4 : K = 2j   + (L>=16 ? 8 : 0)
//   j>=4: K = 16+2(j-4) + (L>=16 ? 8 : 0)
__device__ inline v16h load_frag(const _Float16* base, int stride) {
  int lane = threadIdx.x & 31;
  int r = lane & 15;
  int lo = (lane >> 4) << 2;  // dword offset 0 or 4
  const unsigned* p = (const unsigned*)(base + r * stride);
  union { v16h v; unsigned u[8]; } f;
#pragma unroll
  for (int j = 0; j < 4; ++j) {
    f.u[j]     = p[j + lo];
    f.u[4 + j] = p[8 + j + lo];
  }
  return f.v;
}

// Gather an A fragment directly from global fp32 (row-major, stride floats),
// scaling on the fly and converting to f16.
__device__ inline v16h load_afrag_g(const float* base, int stride, float scale) {
  int lane = threadIdx.x & 31;
  int m = lane & 15;
  int koff = (lane >> 4) << 3;  // 0 or 8
  const float* p = base + (long)m * stride + koff;
  union { v16h v; _Float16 h[16]; } f;
#pragma unroll
  for (int j = 0; j < 8; ++j) {
    int k = (j < 4) ? (2 * j) : (16 + 2 * (j - 4));
    f.h[2 * j]     = (_Float16)(p[k] * scale);
    f.h[2 * j + 1] = (_Float16)(p[k + 1] * scale);
  }
  return f.v;
}

// ---------------------------------------------------------------------------
// fp32 -> f16 weight conversion (grid-stride)
// ---------------------------------------------------------------------------
__global__ void cvt_f32_f16(const float* __restrict__ s, _Float16* __restrict__ d,
                            int n) {
  for (int i = blockIdx.x * blockDim.x + threadIdx.x; i < n;
       i += gridDim.x * blockDim.x)
    d[i] = (_Float16)s[i];
}

// ---------------------------------------------------------------------------
// Generic GEMM: C[M,N] = A[M,K](f32) * W[N,K](f16)^T + bias, optional ReLU.
// Block tile 64x128, BK=32. 256 threads = 8 waves; wave w owns N-strip w,
// computing 4 stacked 16x16 accumulators. Weight tile staged via async
// global->LDS DMA (ASYNCcnt) when available; A tile converted f32->f16 by
// VALU in parallel with the DMA.
// ---------------------------------------------------------------------------
#define GTM 64
#define GTN 128
#define GTK 32

__global__ __launch_bounds__(256) void gemm_f16w(
    const float* __restrict__ A, int lda, const _Float16* __restrict__ W, int ldw,
    float* __restrict__ C, int ldc, const float* __restrict__ bias, int N, int K,
    int relu) {
  __shared__ _Float16 As[GTM][GTK + 8];
  __shared__ _Float16 Bs[GTN][GTK + 8];

  int bm = blockIdx.x * GTM;
  int bn = blockIdx.y * GTN;
  int t = threadIdx.x;
  int wave = t >> 5;
  int lane = t & 31;

  v8f acc[4];
#pragma unroll
  for (int i = 0; i < 4; ++i) acc[i] = zero8();

  int ar = t >> 2;            // 0..63
  int ac = (t & 3) * 8;       // 0,8,16,24
  int br = t >> 1;            // 0..127
  int bc = (t & 1) * 16;      // 0,16

  for (int k0 = 0; k0 < K; k0 += GTK) {
    // stage W tile (already f16, [N][K] row-major): async DMA when available
    {
      const _Float16* wp = W + (long)(bn + br) * ldw + k0 + bc;
#ifdef HAVE_ASYNC_LDS
      async_copy16(wp, &Bs[br][bc]);
      async_copy16(wp + 8, &Bs[br][bc + 8]);
#else
      const v8h* wv = (const v8h*)wp;
      *(v8h*)&Bs[br][bc]     = wv[0];
      *(v8h*)&Bs[br][bc + 8] = wv[1];
#endif
    }
    // stage A tile (fp32 -> f16) — overlaps with the async weight DMA
    {
      const float* ap = A + (long)(bm + ar) * lda + k0 + ac;
      if (k0 + GTK < K) __builtin_prefetch(ap + GTK, 0, 1);
      float4 x0 = *(const float4*)ap;
      float4 x1 = *(const float4*)(ap + 4);
      v8h hv;
      hv[0] = (_Float16)x0.x; hv[1] = (_Float16)x0.y;
      hv[2] = (_Float16)x0.z; hv[3] = (_Float16)x0.w;
      hv[4] = (_Float16)x1.x; hv[5] = (_Float16)x1.y;
      hv[6] = (_Float16)x1.z; hv[7] = (_Float16)x1.w;
      *(v8h*)&As[ar][ac] = hv;
    }
#ifdef HAVE_ASYNC_LDS
    wait_async0();
#endif
    __syncthreads();

    v16h bf = load_frag(&Bs[wave * 16][0], GTK + 8);
#pragma unroll
    for (int mi = 0; mi < 4; ++mi) {
      v16h af = load_frag(&As[mi * 16][0], GTK + 8);
      acc[mi] = wmma_f16(af, bf, acc[mi]);
    }
    __syncthreads();
  }

  int ncol = bn + wave * 16 + (lane & 15);
  int mtop = (lane >> 4) * 8;
  float bv = bias ? bias[ncol] : 0.f;
#pragma unroll
  for (int mi = 0; mi < 4; ++mi) {
#pragma unroll
    for (int r = 0; r < 8; ++r) {
      float v = acc[mi][r] + bv;
      if (relu) v = fmaxf(v, 0.f);
      C[(long)(bm + mi * 16 + mtop + r) * ldc + ncol] = v;
    }
  }
  (void)N;
}

// ---------------------------------------------------------------------------
// Flash attention. One block per (batch, head). 8 waves; wave handles q-tiles
// {wave, wave+8} (16 rows each). K/V streamed in 32-key chunks via LDS.
// O layout: [NTOK, 512] with head offset h*64. Scale 1/8 folded into Q.
// ---------------------------------------------------------------------------
__global__ __launch_bounds__(256) void attention_kernel(
    const float* __restrict__ Q, int qstride, const float* __restrict__ K,
    const float* __restrict__ V, int kvstride, float* __restrict__ O,
    int causal) {
  int blk = blockIdx.x;
  int n = blk >> 3, h = blk & 7;
  const float* qb = Q + (long)n * SEQ * qstride + h * DH;
  const float* kb = K + (long)n * SEQ * kvstride + h * DH;
  const float* vb = V + (long)n * SEQ * kvstride + h * DH;
  float* ob = O + (long)n * SEQ * D_MODEL + h * DH;

  __shared__ _Float16 Ks[32][72];      // [key][d]  (B-frag layout for Q*K^T)
  __shared__ _Float16 Vt[64][40];      // [d][key]  (B-frag layout for P*V)
  __shared__ _Float16 Ps[8][16][36];   // per-wave P staging (A-frag relayout)

  int t = threadIdx.x;
  int wave = t >> 5;
  int lane = t & 31;
  int ncol = lane & 15;
  int mtop = (lane >> 4) * 8;

  v16h a0[2], a1[2];
  v8f o[2][4];
  float rmax[2][8], rsum[2][8];
#pragma unroll
  for (int qs = 0; qs < 2; ++qs) {
    int qt = wave + qs * 8;
    a0[qs] = load_afrag_g(qb + (long)qt * 16 * qstride, qstride, 0.125f);
    a1[qs] = load_afrag_g(qb + (long)qt * 16 * qstride + 32, qstride, 0.125f);
#pragma unroll
    for (int j = 0; j < 4; ++j) o[qs][j] = zero8();
#pragma unroll
    for (int r = 0; r < 8; ++r) { rmax[qs][r] = -1e30f; rsum[qs][r] = 0.f; }
  }

  for (int kc = 0; kc < 8; ++kc) {
    // stage K chunk [32][64]
    {
      int r = t >> 3;          // 0..31
      int c = (t & 7) * 8;     // 0..56
      const float* kp = kb + (long)(kc * 32 + r) * kvstride + c;
#pragma unroll
      for (int j = 0; j < 8; ++j) Ks[r][c + j] = (_Float16)kp[j];
    }
    // stage V chunk transposed -> Vt[d][key]
    for (int idx = t; idx < 2048; idx += 256) {
      int key = idx >> 6, d = idx & 63;
      Vt[d][key] = (_Float16)vb[(long)(kc * 32 + key) * kvstride + d];
    }
    __syncthreads();

#pragma unroll
    for (int qs = 0; qs < 2; ++qs) {
      int qt = wave + qs * 8;
      if (!causal || kc * 32 <= qt * 16 + 15) {   // wave-uniform
        v8f s0 = zero8(), s1 = zero8();
        v16h b0 = load_frag(&Ks[0][0], 72);
        v16h b1 = load_frag(&Ks[0][32], 72);
        s0 = wmma_f16(a0[qs], b0, s0);
        s0 = wmma_f16(a1[qs], b1, s0);
        v16h b2 = load_frag(&Ks[16][0], 72);
        v16h b3 = load_frag(&Ks[16][32], 72);
        s1 = wmma_f16(a0[qs], b2, s1);
        s1 = wmma_f16(a1[qs], b3, s1);

        if (causal) {
#pragma unroll
          for (int r = 0; r < 8; ++r) {
            int q = qt * 16 + mtop + r;
            if (kc * 32 + ncol > q)      s0[r] = -1e30f;
            if (kc * 32 + 16 + ncol > q) s1[r] = -1e30f;
          }
        }
#pragma unroll
        for (int r = 0; r < 8; ++r) {
          float mrow = fmaxf(s0[r], s1[r]);
#pragma unroll
          for (int m = 1; m < 16; m <<= 1)
            mrow = fmaxf(mrow, __shfl_xor(mrow, m, 32));
          float nm = fmaxf(rmax[qs][r], mrow);
          float sf = __expf(rmax[qs][r] - nm);
          rmax[qs][r] = nm;
          float p0 = __expf(s0[r] - nm);
          float p1 = __expf(s1[r] - nm);
          float ps = p0 + p1;
#pragma unroll
          for (int m = 1; m < 16; m <<= 1) ps += __shfl_xor(ps, m, 32);
          rsum[qs][r] = rsum[qs][r] * sf + ps;
#pragma unroll
          for (int j = 0; j < 4; ++j) o[qs][j][r] *= sf;
          Ps[wave][mtop + r][ncol]      = (_Float16)p0;
          Ps[wave][mtop + r][16 + ncol] = (_Float16)p1;
        }
        asm volatile("s_wait_dscnt 0x0" ::: "memory");  // intra-wave LDS RAW
        v16h pf = load_frag(&Ps[wave][0][0], 36);
#pragma unroll
        for (int j = 0; j < 4; ++j) {
          v16h vf = load_frag(&Vt[j * 16][0], 40);
          o[qs][j] = wmma_f16(pf, vf, o[qs][j]);
        }
      }
    }
    __syncthreads();
  }

#pragma unroll
  for (int qs = 0; qs < 2; ++qs) {
    int qt = wave + qs * 8;
#pragma unroll
    for (int r = 0; r < 8; ++r) {
      float inv = 1.f / rsum[qs][r];
#pragma unroll
      for (int j = 0; j < 4; ++j)
        ob[(long)(qt * 16 + mtop + r) * D_MODEL + j * 16 + ncol] =
            o[qs][j][r] * inv;
    }
  }
}

// ---------------------------------------------------------------------------
// out[row] = LN(A[row] + B[row]) * g + beta   (row length 512, in-place safe)
// ---------------------------------------------------------------------------
__global__ __launch_bounds__(256) void ln_res_kernel(
    const float* __restrict__ A, const float* __restrict__ B,
    const float* __restrict__ g, const float* __restrict__ beta,
    float* __restrict__ out) {
  int row = blockIdx.x;
  int t = threadIdx.x;
  const float* pa = A + (long)row * D_MODEL;
  const float* pb = B + (long)row * D_MODEL;
  float v0 = pa[t] + pb[t];
  float v1 = pa[t + 256] + pb[t + 256];
  float s = v0 + v1, sq = v0 * v0 + v1 * v1;
#pragma unroll
  for (int m = 1; m < 32; m <<= 1) {
    s += __shfl_xor(s, m, 32);
    sq += __shfl_xor(sq, m, 32);
  }
  __shared__ float ss[8], ssq[8];
  if ((t & 31) == 0) { ss[t >> 5] = s; ssq[t >> 5] = sq; }
  __syncthreads();
  s = 0.f; sq = 0.f;
#pragma unroll
  for (int i = 0; i < 8; ++i) { s += ss[i]; sq += ssq[i]; }
  float mean = s * (1.f / 512.f);
  float var = sq * (1.f / 512.f) - mean * mean;
  float inv = rsqrtf(var + 1e-5f);
  out[(long)row * D_MODEL + t]       = (v0 - mean) * inv * g[t] + beta[t];
  out[(long)row * D_MODEL + t + 256] = (v1 - mean) * inv * g[t + 256] + beta[t + 256];
}

// x[(n,s),d] += combo_embed[s,d]
__global__ void add_combo(const float* __restrict__ ce, float* __restrict__ x) {
  long i = (long)blockIdx.x * 256 + threadIdx.x;
  int row = (int)(i >> 9);
  int d = (int)(i & 511);
  int s = row & 255;
  x[i] += ce[s * D_MODEL + d];
}

// decoder teacher-forcing input
__global__ __launch_bounds__(128) void dec_embed(
    const float* __restrict__ act_combo, const int* __restrict__ order,
    const float* __restrict__ combo_embed, const float* __restrict__ pos_embed,
    const float* __restrict__ bos, const float* __restrict__ prev_w,
    const float* __restrict__ prev_b, float* __restrict__ tgt) {
  int row = blockIdx.x;
  int n = row >> 8, s = row & 255;
  int t = threadIdx.x;
  __shared__ int idx;
  if (t == 0) {
    int id = -1;
    if (s > 0) {
      int op = order[n * SEQ + (s - 1)];
      const float* ac = act_combo + ((long)n * SEQ + op) * KP1;
      for (int k = 0; k < KP1; ++k)
        if (ac[k] > 0.5f) { id = k; break; }
    }
    idx = id;
  }
  __syncthreads();
  int ocur = order[n * SEQ + s];
  int id = idx;
#pragma unroll
  for (int j = 0; j < 4; ++j) {
    int d = t + j * 128;
    float v = combo_embed[ocur * D_MODEL + d] + pos_embed[s * D_MODEL + d] +
              bos[d] + prev_b[d];
    if (id >= 0) v += prev_w[d * KP1 + id];
    tgt[(long)row * D_MODEL + d] = v;
  }
}

// per-(n,s): selected logp and entropy partial (deterministic)
__global__ __launch_bounds__(128) void row_softmax_stats(
    const float* __restrict__ logits, const float* __restrict__ act_combo,
    const int* __restrict__ order, float* __restrict__ red) {
  int row = blockIdx.x;
  int n = row >> 8, s = row & 255;
  int t = threadIdx.x;
  __shared__ float sm[4], sv[4], sve[4], slp;
  if (t == 0) slp = 0.f;
  float lg = logits[(long)row * KP1 + t];
  float mx = lg;
#pragma unroll
  for (int m = 1; m < 32; m <<= 1) mx = fmaxf(mx, __shfl_xor(mx, m, 32));
  if ((t & 31) == 0) sm[t >> 5] = mx;
  __syncthreads();
  mx = fmaxf(fmaxf(sm[0], sm[1]), fmaxf(sm[2], sm[3]));
  float e = __expf(lg - mx);
  float se = e;
#pragma unroll
  for (int m = 1; m < 32; m <<= 1) se += __shfl_xor(se, m, 32);
  if ((t & 31) == 0) sv[t >> 5] = se;
  __syncthreads();
  se = sv[0] + sv[1] + sv[2] + sv[3];
  float lse = mx + __logf(se);
  float logp = lg - lse;
  float entp = -__expf(logp) * logp;
#pragma unroll
  for (int m = 1; m < 32; m <<= 1) entp += __shfl_xor(entp, m, 32);
  if ((t & 31) == 0) sve[t >> 5] = entp;
  int op = order[n * SEQ + s];
  const float* ac = act_combo + ((long)n * SEQ + op) * KP1;
  if (ac[t] > 0.5f) slp = logp;  // exactly one lane (one-hot)
  __syncthreads();
  if (t == 0) {
    red[row * 2 + 0] = slp;
    red[row * 2 + 1] = sve[0] + sve[1] + sve[2] + sve[3];
  }
}

__global__ __launch_bounds__(256) void final_sum(const float* __restrict__ red,
                                                 float* __restrict__ out) {
  int n = blockIdx.x;
  int t = threadIdx.x;
  float lp = red[(n * SEQ + t) * 2 + 0];
  float en = red[(n * SEQ + t) * 2 + 1];
#pragma unroll
  for (int m = 1; m < 32; m <<= 1) {
    lp += __shfl_xor(lp, m, 32);
    en += __shfl_xor(en, m, 32);
  }
  __shared__ float a[8], b[8];
  if ((t & 31) == 0) { a[t >> 5] = lp; b[t >> 5] = en; }
  __syncthreads();
  if (t == 0) {
    float L = 0.f, E = 0.f;
#pragma unroll
    for (int i = 0; i < 8; ++i) { L += a[i]; E += b[i]; }
    out[n] = L;
    out[NBATCH + n] = E;
  }
}

// ---------------------------------------------------------------------------
// Host orchestration
// ---------------------------------------------------------------------------
extern "C" void kernel_launch(void* const* d_in, const int* in_sizes, int n_in,
                              void* d_out, int out_size, void* d_ws,
                              size_t ws_size, hipStream_t stream) {
  (void)in_sizes; (void)n_in; (void)out_size; (void)ws_size;
  // --- inputs (setup_inputs dict insertion order, params dict recursively) ---
  const float* obs_belt   = (const float*)d_in[0];
  const float* act_combo  = (const float*)d_in[1];
  const float* obs_proj_w = (const float*)d_in[2];
  const float* obs_proj_b = (const float*)d_in[3];
  const float* combo_emb  = (const float*)d_in[4];
  const float* pos_emb    = (const float*)d_in[5];
  const float* bos        = (const float*)d_in[6];
  const float* prev_w     = (const float*)d_in[7];
  const float* prev_b     = (const float*)d_in[8];
  const float* head_w     = (const float*)d_in[9];
  const float* head_b     = (const float*)d_in[10];
  const float* enc_qkv_w  = (const float*)d_in[11];
  const float* enc_qkv_b  = (const float*)d_in[12];
  const float* enc_o_w    = (const float*)d_in[13];
  const float* enc_o_b    = (const float*)d_in[14];
  const float* enc_ff1_w  = (const float*)d_in[15];
  const float* enc_ff1_b  = (const float*)d_in[16];
  const float* enc_ff2_w  = (const float*)d_in[17];
  const float* enc_ff2_b  = (const float*)d_in[18];
  const float* enc_ln1_g  = (const float*)d_in[19];
  const float* enc_ln1_b  = (const float*)d_in[20];
  const float* enc_ln2_g  = (const float*)d_in[21];
  const float* enc_ln2_b  = (const float*)d_in[22];
  const float* dsa_qkv_w  = (const float*)d_in[23];
  const float* dsa_qkv_b  = (const float*)d_in[24];
  const float* dsa_o_w    = (const float*)d_in[25];
  const float* dsa_o_b    = (const float*)d_in[26];
  const float* dca_qkv_w  = (const float*)d_in[27];
  const float* dca_qkv_b  = (const float*)d_in[28];
  const float* dca_o_w    = (const float*)d_in[29];
  const float* dca_o_b    = (const float*)d_in[30];
  const float* dff1_w     = (const float*)d_in[31];
  const float* dff1_b     = (const float*)d_in[32];
  const float* dff2_w     = (const float*)d_in[33];
  const float* dff2_b     = (const float*)d_in[34];
  const float* dln1_g     = (const float*)d_in[35];
  const float* dln1_b     = (const float*)d_in[36];
  const float* dln2_g     = (const float*)d_in[37];
  const float* dln2_b     = (const float*)d_in[38];
  const float* dln3_g     = (const float*)d_in[39];
  const float* dln3_b     = (const float*)d_in[40];
  const int*   order      = (const int*)d_in[41];

  // --- workspace layout ---
  char* ws = (char*)d_ws;
  _Float16* wh = (_Float16*)ws;                       // 29,458,432 halves
  float* xbuf   = (float*)(ws + 58916864);            // [16384,512] (enc x / memory)
  float* ybuf   = xbuf + 8388608;                     // [16384,512] (dec y)
  float* attnA  = ybuf + 8388608;                     // [16384,512]
  float* tmpB   = attnA + 8388608;                    // [16384,512]
  float* qkvbuf = tmpB + 8388608;                     // [16384,1536]
  float* big2   = qkvbuf + 25165824;                  // [16384,2048] hidden/logits
  float* redbuf = big2 + 33554432;                    // [16384,2]

  // f16 weight arena offsets (halves)
  const long wo_obs     = 0;
  const long wo_head    = 32768;
  const long wo_eqkv    = 98304;
  const long wo_eo      = 3244032;
  const long wo_eff1    = 4292608;
  const long wo_eff2    = 8486912;
  const long wo_dsaqkv  = 12681216;
  const long wo_dsao    = 15826944;
  const long wo_dcaqkv  = 16875520;
  const long wo_dcao    = 20021248;
  const long wo_dff1    = 21069824;
  const long wo_dff2    = 25264128;

  auto cvt = [&](const float* src, long off, int count) {
    cvt_f32_f16<<<dim3(1024), dim3(256), 0, stream>>>(src, wh + off, count);
  };
  cvt(obs_proj_w, wo_obs, 512 * 64);
  cvt(head_w, wo_head, 128 * 512);
  cvt(enc_qkv_w, wo_eqkv, 4 * 1536 * 512);
  cvt(enc_o_w, wo_eo, 4 * 512 * 512);
  cvt(enc_ff1_w, wo_eff1, 4 * 2048 * 512);
  cvt(enc_ff2_w, wo_eff2, 4 * 512 * 2048);
  cvt(dsa_qkv_w, wo_dsaqkv, 4 * 1536 * 512);
  cvt(dsa_o_w, wo_dsao, 4 * 512 * 512);
  cvt(dca_qkv_w, wo_dcaqkv, 4 * 1536 * 512);
  cvt(dca_o_w, wo_dcao, 4 * 512 * 512);
  cvt(dff1_w, wo_dff1, 4 * 2048 * 512);
  cvt(dff2_w, wo_dff2, 4 * 512 * 2048);

  auto gemm = [&](const float* A, int lda, long woff, int ldw, float* C, int ldc,
                  const float* bias, int N, int K, int relu) {
    dim3 g(NTOK / GTM, N / GTN);
    gemm_f16w<<<g, dim3(256), 0, stream>>>(A, lda, wh + woff, ldw, C, ldc, bias,
                                           N, K, relu);
  };
  auto attn = [&](const float* Q, int qs, const float* K, const float* V, int kvs,
                  float* O, int causal) {
    attention_kernel<<<dim3(NBATCH * NHEAD), dim3(256), 0, stream>>>(
        Q, qs, K, V, kvs, O, causal);
  };
  auto lnres = [&](const float* A, const float* B, const float* g,
                   const float* b, float* out) {
    ln_res_kernel<<<dim3(NTOK), dim3(256), 0, stream>>>(A, B, g, b, out);
  };

  // ================= encoder =================
  gemm(obs_belt, OBS_DIM, wo_obs, OBS_DIM, xbuf, D_MODEL, obs_proj_b, D_MODEL,
       OBS_DIM, 0);
  add_combo<<<dim3(NTOK * D_MODEL / 256), dim3(256), 0, stream>>>(combo_emb, xbuf);

  for (int i = 0; i < NLAYERS; ++i) {
    gemm(xbuf, D_MODEL, wo_eqkv + (long)i * 1536 * 512, D_MODEL, qkvbuf, 1536,
         enc_qkv_b + i * 1536, 1536, D_MODEL, 0);
    attn(qkvbuf, 1536, qkvbuf + 512, qkvbuf + 1024, 1536, attnA, 0);
    gemm(attnA, D_MODEL, wo_eo + (long)i * 512 * 512, D_MODEL, tmpB, D_MODEL,
         enc_o_b + i * 512, D_MODEL, D_MODEL, 0);
    lnres(xbuf, tmpB, enc_ln1_g + i * 512, enc_ln1_b + i * 512, xbuf);
    gemm(xbuf, D_MODEL, wo_eff1 + (long)i * 2048 * 512, D_MODEL, big2, FFN,
         enc_ff1_b + i * 2048, FFN, D_MODEL, 1);
    gemm(big2, FFN, wo_eff2 + (long)i * 512 * 2048, FFN, tmpB, D_MODEL,
         enc_ff2_b + i * 512, D_MODEL, FFN, 0);
    lnres(xbuf, tmpB, enc_ln2_g + i * 512, enc_ln2_b + i * 512, xbuf);
  }
  // xbuf now holds `memory`

  // ================= decoder =================
  dec_embed<<<dim3(NTOK), dim3(128), 0, stream>>>(act_combo, order, combo_emb,
                                                  pos_emb, bos, prev_w, prev_b,
                                                  ybuf);
  float* qbuf = qkvbuf;
  float* kvbuf = qkvbuf + (long)NTOK * 512;

  for (int i = 0; i < NLAYERS; ++i) {
    // masked self-attention
    gemm(ybuf, D_MODEL, wo_dsaqkv + (long)i * 1536 * 512, D_MODEL, qkvbuf, 1536,
         dsa_qkv_b + i * 1536, 1536, D_MODEL, 0);
    attn(qkvbuf, 1536, qkvbuf + 512, qkvbuf + 1024, 1536, attnA, 1);
    gemm(attnA, D_MODEL, wo_dsao + (long)i * 512 * 512, D_MODEL, tmpB, D_MODEL,
         dsa_o_b + i * 512, D_MODEL, D_MODEL, 0);
    lnres(ybuf, tmpB, dln1_g + i * 512, dln1_b + i * 512, ybuf);
    // cross-attention: Q from y, K/V from memory
    gemm(ybuf, D_MODEL, wo_dcaqkv + (long)i * 1536 * 512, D_MODEL, qbuf, 512,
         dca_qkv_b + i * 1536, 512, D_MODEL, 0);
    gemm(xbuf, D_MODEL, wo_dcaqkv + (long)i * 1536 * 512 + 512 * 512, D_MODEL,
         kvbuf, 1024, dca_qkv_b + i * 1536 + 512, 1024, D_MODEL, 0);
    attn(qbuf, 512, kvbuf, kvbuf + 512, 1024, attnA, 0);
    gemm(attnA, D_MODEL, wo_dcao + (long)i * 512 * 512, D_MODEL, tmpB, D_MODEL,
         dca_o_b + i * 512, D_MODEL, D_MODEL, 0);
    lnres(ybuf, tmpB, dln2_g + i * 512, dln2_b + i * 512, ybuf);
    // FFN
    gemm(ybuf, D_MODEL, wo_dff1 + (long)i * 2048 * 512, D_MODEL, big2, FFN,
         dff1_b + i * 2048, FFN, D_MODEL, 1);
    gemm(big2, FFN, wo_dff2 + (long)i * 512 * 2048, FFN, tmpB, D_MODEL,
         dff2_b + i * 512, D_MODEL, FFN, 0);
    lnres(ybuf, tmpB, dln3_g + i * 512, dln3_b + i * 512, ybuf);
  }

  // ================= head + loss stats =================
  gemm(ybuf, D_MODEL, wo_head, D_MODEL, big2, KP1, head_b, KP1, D_MODEL, 0);
  row_softmax_stats<<<dim3(NTOK), dim3(128), 0, stream>>>(big2, act_combo, order,
                                                          redbuf);
  final_sum<<<dim3(NBATCH), dim3(256), 0, stream>>>(redbuf, (float*)d_out);
}
